// ProjectedConjugatedCSPNet_46239617909187
// MI455X (gfx1250) — matrix operand order repaired and checked by reading, not compiled, not run on hardware
//
#include <hip/hip_runtime.h>
#include <hip/hip_bf16.h>
#include <math.h>

// ---------------------------------------------------------------------------
// Fused DiffCSP-style message-passing layer for MI455X (gfx1250, wave32).
// All GEMMs via v_wmma_f32_16x16x32_bf16. Edge features never touch HBM:
// they are scatter-added into agg with f32 atomics straight from registers.
// Wave assignment: each wave owns a unique group of 4 N-tiles and walks all
// 4 M-tiles, so every B fragment feeds 4 WMMAs and no B loads are duplicated
// within a block (4x less L2 weight traffic than the naive mapping).
// ---------------------------------------------------------------------------

typedef __attribute__((ext_vector_type(16))) __bf16 v16bf;
typedef __attribute__((ext_vector_type(8)))  float  v8f;

#define H        512
#define KP_E1    1824   // 1798 padded to a multiple of 32
#define K_EIN    1798
#define KP_N1    1024
#define FD0      1024   // start of lattice+sinusoid region in e_in
#define FDLEN    800    // KP_E1 - FD0 (staged in LDS)
#define BM       64     // rows (edges / nodes) per block
#define NTHREADS 256    // 8 waves

__device__ __forceinline__ unsigned short f32_to_bf16(float f) {
  unsigned int u = __float_as_uint(f);
  unsigned int r = u + 0x7FFFu + ((u >> 16) & 1u);  // round-to-nearest-even
  return (unsigned short)(r >> 16);
}

__device__ __forceinline__ float silu(float x) {
  return x / (1.0f + __expf(-x));
}

// A-fragment: 16x32 bf16 tile, row = lane%16, from a row pointer with
// arbitrary (16B-aligned) element stride already applied by the caller.
// lane<16 : elements 0..15 = K kb..kb+15
// lane>=16: elements 0..15 = K kb+16..kb+31
__device__ __forceinline__ v16bf load_a_frag(const unsigned short* rowp, int kb, int hi) {
  union { v16bf v; uint4 q[2]; } u;
  u.q[0] = *(const uint4*)(rowp + kb + hi * 16);
  u.q[1] = *(const uint4*)(rowp + kb + hi * 16 + 8);
  return u.v;
}

// B-fragment: 32x16 bf16 tile from transposed weights WT[n][Kpad].
// lane holds column n = base + lane%16, 16 contiguous K from kb + 16*(lane/16).
__device__ __forceinline__ v16bf load_b_frag(const unsigned short* wt, int n, int Kpad,
                                             int kb, int hi) {
  union { v16bf v; uint4 q[2]; } u;
  const unsigned short* p = wt + (size_t)n * Kpad + kb + hi * 16;
  u.q[0] = *(const uint4*)(p);
  u.q[1] = *(const uint4*)(p + 8);
  return u.v;
}

// ---------------------------------------------------------------------------
// LayerNorm -> bf16
// ---------------------------------------------------------------------------
__global__ void ln_kernel(const float* __restrict__ x, const float* __restrict__ g,
                          const float* __restrict__ b, unsigned short* __restrict__ h,
                          int N) {
  int row = blockIdx.x;
  int tid = threadIdx.x;                      // 256 threads, 2 elems each
  float2 v = *(const float2*)(x + (size_t)row * H + tid * 2);
  float s  = v.x + v.y;
  float ss = v.x * v.x + v.y * v.y;
  #pragma unroll
  for (int m = 16; m >= 1; m >>= 1) {
    s  += __shfl_xor(s,  m, 32);
    ss += __shfl_xor(ss, m, 32);
  }
  __shared__ float ps[8], pss[8];
  int wave = tid >> 5, lane = tid & 31;
  if (lane == 0) { ps[wave] = s; pss[wave] = ss; }
  __syncthreads();
  if (tid == 0) {
    float a = 0.f, c = 0.f;
    #pragma unroll
    for (int w = 0; w < 8; ++w) { a += ps[w]; c += pss[w]; }
    ps[0]  = a * (1.0f / (float)H);
    pss[0] = c * (1.0f / (float)H);
  }
  __syncthreads();
  float mean = ps[0];
  float var  = pss[0] - mean * mean;
  float rstd = rsqrtf(var + 1e-5f);
  int k0 = tid * 2;
  float y0 = (v.x - mean) * rstd * g[k0]     + b[k0];
  float y1 = (v.y - mean) * rstd * g[k0 + 1] + b[k0 + 1];
  h[(size_t)row * H + k0]     = f32_to_bf16(y0);
  h[(size_t)row * H + k0 + 1] = f32_to_bf16(y1);
}

// ---------------------------------------------------------------------------
// Weight transpose fp32 [K][Nw] -> bf16 [Nw][Kpad] (zero-padded K)
// ---------------------------------------------------------------------------
__global__ void wt_transpose_kernel(const float* __restrict__ W,
                                    unsigned short* __restrict__ WT,
                                    int K, int Nw, int Kpad) {
  int i = blockIdx.x * blockDim.x + threadIdx.x;
  int total = Nw * Kpad;
  if (i >= total) return;
  int n = i / Kpad, k = i - n * Kpad;
  float v = (k < K) ? W[(size_t)k * Nw + n] : 0.0f;
  WT[(size_t)n * Kpad + k] = f32_to_bf16(v);
}

__global__ void zero_kernel(float* __restrict__ p, long long n) {
  long long i = (long long)blockIdx.x * blockDim.x + threadIdx.x;
  if (i < n) p[i] = 0.0f;
}

// ---------------------------------------------------------------------------
// Edge kernel (BM=64 edges per block).
// A operand split: K<1024 (gathered h rows) read straight from global/L2;
// K 1024..1823 (lattice + sin/cos embedding, computed) staged in LDS.
// ---------------------------------------------------------------------------
__global__ void __launch_bounds__(NTHREADS)
edge_kernel(const unsigned short* __restrict__ h_bf,
            const float* __restrict__ lattices,
            const int* __restrict__ edge_index,      // [2,E]
            const int* __restrict__ edge2graph,
            const float* __restrict__ frac_diff,     // [E,3]
            const unsigned short* __restrict__ We1T, // [512][1824] bf16
            const float* __restrict__ be1,
            const unsigned short* __restrict__ We2T, // [512][512] bf16
            const float* __restrict__ be2,
            float* __restrict__ agg, float* __restrict__ cnt,
            int E) {
  __shared__ __align__(32) unsigned short smF[BM * FDLEN];  // 100 KB; reused as e1 [64][512]
  __shared__ int s_src[BM];
  __shared__ int s_dst[BM];

  int tid = threadIdx.x;
  int e0  = blockIdx.x * BM;

  if (tid < BM) {
    int e = e0 + tid;
    s_src[tid] = (e < E) ? edge_index[e] : 0;
    s_dst[tid] = (e < E) ? edge_index[E + e] : 0;
  }

  // ---- stage [lat(6) | sin(384) cos(384) | pad] region (K 1024..1823) ----
  for (int r = 0; r < BM; ++r) {
    int e = e0 + r;
    unsigned short* row = smF + r * FDLEN;
    if (e < E) {
      int gph  = edge2graph[e];
      float x0 = frac_diff[(size_t)e * 3 + 0];
      float x1 = frac_diff[(size_t)e * 3 + 1];
      float x2 = frac_diff[(size_t)e * 3 + 2];
      for (int k = FD0 + tid; k < KP_E1; k += NTHREADS) {
        unsigned short val;
        if (k < FD0 + 6) {
          val = f32_to_bf16(lattices[gph * 6 + (k - FD0)]);
        } else if (k < K_EIN) {
          int j  = k - (FD0 + 6);                   // 0..767
          int jj = (j < 384) ? j : j - 384;         // space*128 + freq
          float xs  = (jj < 128) ? x0 : ((jj < 256) ? x1 : x2);
          float ang = xs * (6.283185307179586f * (float)(jj & 127));
          val = f32_to_bf16((j < 384) ? __sinf(ang) : __cosf(ang));
        } else {
          val = 0;   // pad 1798..1823
        }
        row[k - FD0] = val;
      }
    } else {
      for (int k = tid; k < FDLEN; k += NTHREADS) row[k] = 0;
    }
  }
  __syncthreads();

  int w    = tid >> 5;          // wave 0..7: owns N-tiles 4w..4w+3
  int lane = tid & 31;
  int hi   = lane >> 4;
  int lm   = lane & 15;

  // per-lane A row pointers (gathered h rows) for each M-tile
  const unsigned short* psrc[4];
  const unsigned short* pdst[4];
  #pragma unroll
  for (int m = 0; m < 4; ++m) {
    int row = m * 16 + lm;
    psrc[m] = h_bf + (size_t)s_src[row] * H;
    pdst[m] = h_bf + (size_t)s_dst[row] * H;
  }

  // ---- GEMM 1: [64 x 1824] x [1824 x 512], acc[4 Mtiles][4 Ntiles] ----
  v8f acc[4][4] = {};
  // phase 1: K 0..511 (h[src], from global)
  for (int kb = 0; kb < H; kb += 32) {
    v16bf a[4];
    #pragma unroll
    for (int m = 0; m < 4; ++m) a[m] = load_a_frag(psrc[m], kb, hi);
    #pragma unroll
    for (int j = 0; j < 4; ++j) {
      v16bf bfr = load_b_frag(We1T, w * 64 + j * 16 + lm, KP_E1, kb, hi);
      #pragma unroll
      for (int m = 0; m < 4; ++m)
        acc[m][j] = __builtin_amdgcn_wmma_f32_16x16x32_bf16(
            false, a[m], false, bfr, (short)0, acc[m][j], false, false);
    }
  }
  // phase 2: K 512..1023 (h[dst], from global)
  for (int kb = 0; kb < H; kb += 32) {
    v16bf a[4];
    #pragma unroll
    for (int m = 0; m < 4; ++m) a[m] = load_a_frag(pdst[m], kb, hi);
    #pragma unroll
    for (int j = 0; j < 4; ++j) {
      v16bf bfr = load_b_frag(We1T, w * 64 + j * 16 + lm, KP_E1, H + kb, hi);
      #pragma unroll
      for (int m = 0; m < 4; ++m)
        acc[m][j] = __builtin_amdgcn_wmma_f32_16x16x32_bf16(
            false, a[m], false, bfr, (short)0, acc[m][j], false, false);
    }
  }
  // phase 3: K 1024..1823 (lat+fd, from LDS)
  for (int kb = 0; kb < FDLEN; kb += 32) {
    v16bf a[4];
    #pragma unroll
    for (int m = 0; m < 4; ++m)
      a[m] = load_a_frag(smF + (m * 16 + lm) * FDLEN, kb, hi);
    #pragma unroll
    for (int j = 0; j < 4; ++j) {
      v16bf bfr = load_b_frag(We1T, w * 64 + j * 16 + lm, KP_E1, FD0 + kb, hi);
      #pragma unroll
      for (int m = 0; m < 4; ++m)
        acc[m][j] = __builtin_amdgcn_wmma_f32_16x16x32_bf16(
            false, a[m], false, bfr, (short)0, acc[m][j], false, false);
    }
  }
  __syncthreads();   // all waves done reading smF; reuse it for e1

  // ---- bias + SiLU -> e1 (bf16, [64][512]) in LDS ----
  unsigned short* e1 = smF;
  #pragma unroll
  for (int j = 0; j < 4; ++j) {
    int col = w * 64 + j * 16 + lm;
    float bia = be1[col];
    #pragma unroll
    for (int m = 0; m < 4; ++m) {
      #pragma unroll
      for (int i = 0; i < 8; ++i) {
        int rrow = m * 16 + i + hi * 8;
        e1[rrow * H + col] = f32_to_bf16(silu(acc[m][j][i] + bia));
      }
    }
  }
  __syncthreads();

  // ---- GEMM 2: [64 x 512] x [512 x 512] ----
  v8f acc2[4][4] = {};
  for (int kb = 0; kb < H; kb += 32) {
    v16bf a[4];
    #pragma unroll
    for (int m = 0; m < 4; ++m)
      a[m] = load_a_frag(e1 + (m * 16 + lm) * H, kb, hi);
    #pragma unroll
    for (int j = 0; j < 4; ++j) {
      v16bf bfr = load_b_frag(We2T, w * 64 + j * 16 + lm, H, kb, hi);
      #pragma unroll
      for (int m = 0; m < 4; ++m)
        acc2[m][j] = __builtin_amdgcn_wmma_f32_16x16x32_bf16(
            false, a[m], false, bfr, (short)0, acc2[m][j], false, false);
    }
  }

  // ---- bias + SiLU + scatter-add into agg ----
  #pragma unroll
  for (int j = 0; j < 4; ++j) {
    int col = w * 64 + j * 16 + lm;
    float bia = be2[col];
    #pragma unroll
    for (int m = 0; m < 4; ++m) {
      #pragma unroll
      for (int i = 0; i < 8; ++i) {
        int rrow = m * 16 + i + hi * 8;
        if (e0 + rrow < E) {
          float v = silu(acc2[m][j][i] + bia);
          atomicAdd(&agg[(size_t)s_src[rrow] * H + col], v);
        }
      }
    }
  }
  if (tid < BM && (e0 + tid) < E) atomicAdd(&cnt[s_src[tid]], 1.0f);
}

// ---------------------------------------------------------------------------
// Node kernel (BM=64 nodes per block): h half of A read straight from global
// (contiguous rows), agg/cnt half staged in LDS (needs the mean division).
// ---------------------------------------------------------------------------
__global__ void __launch_bounds__(NTHREADS)
node_kernel(const unsigned short* __restrict__ h_bf,
            const float* __restrict__ agg, const float* __restrict__ cnt,
            const unsigned short* __restrict__ Wn1T,  // [512][1024] bf16
            const float* __restrict__ bn1,
            const unsigned short* __restrict__ Wn2T,  // [512][512] bf16
            const float* __restrict__ bn2,
            const float* __restrict__ node_features,
            float* __restrict__ out, int N) {
  __shared__ __align__(32) unsigned short smB[BM * H];   // 64 KB; agg half, then e1

  int tid = threadIdx.x;
  int n0  = blockIdx.x * BM;

  for (int r = 0; r < BM; ++r) {
    int node = n0 + r;
    unsigned short* row = smB + r * H;
    if (node < N) {
      float inv = 1.0f / fmaxf(cnt[node], 1.0f);
      for (int k = tid; k < H; k += NTHREADS)
        row[k] = f32_to_bf16(agg[(size_t)node * H + k] * inv);
    } else {
      for (int k = tid; k < H; k += NTHREADS) row[k] = 0;
    }
  }
  __syncthreads();

  int w    = tid >> 5;
  int lane = tid & 31;
  int hi   = lane >> 4;
  int lm   = lane & 15;

  const unsigned short* ph[4];
  #pragma unroll
  for (int m = 0; m < 4; ++m) {
    int node = n0 + m * 16 + lm;
    if (node >= N) node = N - 1;        // keep reads in-bounds; stores are guarded
    ph[m] = h_bf + (size_t)node * H;
  }

  // ---- GEMM 1: [64 x 1024] x [1024 x 512] ----
  v8f acc[4][4] = {};
  // phase 1: K 0..511 (h, global)
  for (int kb = 0; kb < H; kb += 32) {
    v16bf a[4];
    #pragma unroll
    for (int m = 0; m < 4; ++m) a[m] = load_a_frag(ph[m], kb, hi);
    #pragma unroll
    for (int j = 0; j < 4; ++j) {
      v16bf bfr = load_b_frag(Wn1T, w * 64 + j * 16 + lm, KP_N1, kb, hi);
      #pragma unroll
      for (int m = 0; m < 4; ++m)
        acc[m][j] = __builtin_amdgcn_wmma_f32_16x16x32_bf16(
            false, a[m], false, bfr, (short)0, acc[m][j], false, false);
    }
  }
  // phase 2: K 512..1023 (agg mean, LDS)
  for (int kb = 0; kb < H; kb += 32) {
    v16bf a[4];
    #pragma unroll
    for (int m = 0; m < 4; ++m)
      a[m] = load_a_frag(smB + (m * 16 + lm) * H, kb, hi);
    #pragma unroll
    for (int j = 0; j < 4; ++j) {
      v16bf bfr = load_b_frag(Wn1T, w * 64 + j * 16 + lm, KP_N1, H + kb, hi);
      #pragma unroll
      for (int m = 0; m < 4; ++m)
        acc[m][j] = __builtin_amdgcn_wmma_f32_16x16x32_bf16(
            false, a[m], false, bfr, (short)0, acc[m][j], false, false);
    }
  }
  __syncthreads();

  unsigned short* e1 = smB;   // reuse LDS as [64][512]
  #pragma unroll
  for (int j = 0; j < 4; ++j) {
    int col = w * 64 + j * 16 + lm;
    float bia = bn1[col];
    #pragma unroll
    for (int m = 0; m < 4; ++m) {
      #pragma unroll
      for (int i = 0; i < 8; ++i) {
        int rrow = m * 16 + i + hi * 8;
        e1[rrow * H + col] = f32_to_bf16(silu(acc[m][j][i] + bia));
      }
    }
  }
  __syncthreads();

  // ---- GEMM 2: [64 x 512] x [512 x 512] ----
  v8f acc2[4][4] = {};
  for (int kb = 0; kb < H; kb += 32) {
    v16bf a[4];
    #pragma unroll
    for (int m = 0; m < 4; ++m)
      a[m] = load_a_frag(e1 + (m * 16 + lm) * H, kb, hi);
    #pragma unroll
    for (int j = 0; j < 4; ++j) {
      v16bf bfr = load_b_frag(Wn2T, w * 64 + j * 16 + lm, H, kb, hi);
      #pragma unroll
      for (int m = 0; m < 4; ++m)
        acc2[m][j] = __builtin_amdgcn_wmma_f32_16x16x32_bf16(
            false, a[m], false, bfr, (short)0, acc2[m][j], false, false);
    }
  }

  // ---- bias + SiLU + residual ----
  #pragma unroll
  for (int j = 0; j < 4; ++j) {
    int col = w * 64 + j * 16 + lm;
    float bia = bn2[col];
    #pragma unroll
    for (int m = 0; m < 4; ++m) {
      #pragma unroll
      for (int i = 0; i < 8; ++i) {
        int rrow = m * 16 + i + hi * 8;
        int node = n0 + rrow;
        if (node < N) {
          float v = silu(acc2[m][j][i] + bia);
          out[(size_t)node * H + col] = node_features[(size_t)node * H + col] + v;
        }
      }
    }
  }
}

// ---------------------------------------------------------------------------
extern "C" void kernel_launch(void* const* d_in, const int* in_sizes, int n_in,
                              void* d_out, int out_size, void* d_ws, size_t ws_size,
                              hipStream_t stream) {
  const float* node_features = (const float*)d_in[0];
  const float* lattices      = (const float*)d_in[1];
  const int*   edge_index    = (const int*)d_in[2];
  const int*   edge2graph    = (const int*)d_in[3];
  const float* frac_diff     = (const float*)d_in[4];
  // d_in[5] num_atoms: unused by reference math
  const float* ln_g = (const float*)d_in[6];
  const float* ln_b = (const float*)d_in[7];
  const float* W_e1 = (const float*)d_in[8];
  const float* b_e1 = (const float*)d_in[9];
  const float* W_e2 = (const float*)d_in[10];
  const float* b_e2 = (const float*)d_in[11];
  const float* W_n1 = (const float*)d_in[12];
  const float* b_n1 = (const float*)d_in[13];
  const float* W_n2 = (const float*)d_in[14];
  const float* b_n2 = (const float*)d_in[15];

  const int N = in_sizes[0] / H;
  const int E = in_sizes[3];

  // workspace carve (~35 MB)
  size_t off = 0;
  auto carve = [&](size_t bytes) -> void* {
    void* p = (char*)d_ws + off;
    off += (bytes + 255) & ~(size_t)255;
    return p;
  };
  unsigned short* h_bf  = (unsigned short*)carve((size_t)N * H * 2);
  float*          agg   = (float*)carve(((size_t)N * H + N) * 4);
  float*          cnt   = agg + (size_t)N * H;
  unsigned short* We1T  = (unsigned short*)carve((size_t)H * KP_E1 * 2);
  unsigned short* We2T  = (unsigned short*)carve((size_t)H * H * 2);
  unsigned short* Wn1T  = (unsigned short*)carve((size_t)H * KP_N1 * 2);
  unsigned short* Wn2T  = (unsigned short*)carve((size_t)H * H * 2);

  // 1) LayerNorm -> bf16
  ln_kernel<<<N, NTHREADS, 0, stream>>>(node_features, ln_g, ln_b, h_bf, N);

  // 2) weight transposes -> bf16 (L2-resident afterwards)
  {
    int t = H * KP_E1;
    wt_transpose_kernel<<<(t + 255) / 256, 256, 0, stream>>>(W_e1, We1T, K_EIN, H, KP_E1);
    t = H * H;
    wt_transpose_kernel<<<(t + 255) / 256, 256, 0, stream>>>(W_e2, We2T, H, H, H);
    t = H * KP_N1;
    wt_transpose_kernel<<<(t + 255) / 256, 256, 0, stream>>>(W_n1, Wn1T, KP_N1, H, KP_N1);
    t = H * H;
    wt_transpose_kernel<<<(t + 255) / 256, 256, 0, stream>>>(W_n2, Wn2T, H, H, H);
  }

  // 3) zero agg + cnt (contiguous)
  {
    long long t = (long long)N * H + N;
    zero_kernel<<<(unsigned)((t + 255) / 256), 256, 0, stream>>>(agg, t);
  }

  // 4) fused edge MLP + scatter-add
  edge_kernel<<<(E + BM - 1) / BM, NTHREADS, 0, stream>>>(
      h_bf, lattices, edge_index, edge2graph, frac_diff,
      We1T, b_e1, We2T, b_e2, agg, cnt, E);

  // 5) fused node MLP + residual
  node_kernel<<<(N + BM - 1) / BM, NTHREADS, 0, stream>>>(
      h_bf, agg, cnt, Wn1T, b_n1, Wn2T, b_n2, node_features, (float*)d_out, N);
}
